// GATLayer_2267742732743
// MI455X (gfx1250) — matrix operand order repaired and checked by reading, not compiled, not run on hardware
//
#include <hip/hip_runtime.h>
#include <hip/hip_bf16.h>
#include <math.h>

// ---------------------------------------------------------------------------
// GAT layer for MI455X (gfx1250, wave32, WMMA).
// Edge GEMMs ([E x 96]x[96 x 64], [E x 160]x[160 x 32]) on
// v_wmma_f32_16x16x32_bf16 with fp32 accumulate. Operands fetched as b128
// loads: A rows are contiguous 8-element bf16 runs per lane; B (weights) is
// pre-packed into WMMA lane layout. Two M-tiles (32 edges) per wave share
// each B fetch. Segment softmax via global atomics (node tables L2-resident).
// ---------------------------------------------------------------------------

typedef __attribute__((ext_vector_type(16))) __bf16 bf16x16;
typedef __attribute__((ext_vector_type(8)))  __bf16 bf16x8;
typedef __attribute__((ext_vector_type(8)))  float  f32x8;

union BF16Vec {
    bf16x16 v;
    bf16x8  h[2];
};

__device__ __forceinline__ bf16x8 load_bf8(const __bf16* p) {
    return *(const bf16x8*)p;                     // 16-byte load
}
__device__ __forceinline__ bf16x8 cvt_bf8(const float* p) {
    float4 f0 = *(const float4*)p;                // b128
    float4 f1 = *(const float4*)(p + 4);          // b128
    bf16x8 r;
    r[0] = (__bf16)f0.x; r[1] = (__bf16)f0.y; r[2] = (__bf16)f0.z; r[3] = (__bf16)f0.w;
    r[4] = (__bf16)f1.x; r[5] = (__bf16)f1.y; r[6] = (__bf16)f1.z; r[7] = (__bf16)f1.w;
    return r;
}

// ordered-uint encoding so unsigned atomicMax implements float max
__device__ __forceinline__ unsigned f32ord(float f) {
    union { float f; unsigned u; } x; x.f = f;
    return (x.u & 0x80000000u) ? ~x.u : (x.u | 0x80000000u);
}
__device__ __forceinline__ float ordf32(unsigned o) {
    union { unsigned u; float f; } x;
    x.u = (o & 0x80000000u) ? (o & 0x7FFFFFFFu) : ~o;
    return x.f;
}

// ---------------------------------------------------------------------------
// K0: zero h_agg (in d_out) and node max/sum/indeg tables
// ---------------------------------------------------------------------------
__global__ void init_kernel(float* __restrict__ hagg, unsigned* __restrict__ nmax,
                            float* __restrict__ nsum, float* __restrict__ ndeg,
                            int nNodes) {
    int i = blockIdx.x * blockDim.x + threadIdx.x;
    if (i < nNodes * 64) hagg[i] = 0.0f;
    if (i < nNodes) { nmax[i] = 0u; nsum[i] = 0.0f; ndeg[i] = 0.0f; }
}

// ---------------------------------------------------------------------------
// K0b: pack a fp32 weight matrix [K x Nout] into WMMA-B lane layout, bf16.
// Group g = (ks*nt + t)*32 + lane holds 16 contiguous bf16:
//   element i = W[ks*32 + (i<8 ? i : i+8) + (lane>=16 ? 8 : 0)][t*16 + (lane&15)]
// ---------------------------------------------------------------------------
__global__ void pack_weights(const float* __restrict__ W, __bf16* __restrict__ dst,
                             int K, int Nout) {
    int idx = blockIdx.x * blockDim.x + threadIdx.x;
    int nt = Nout >> 4;
    int total = (K >> 5) * nt * 32;
    if (idx >= total) return;
    int lane = idx & 31;
    int t    = (idx >> 5) % nt;
    int ks   = (idx >> 5) / nt;
    int n    = t * 16 + (lane & 15);
    int kofs = ks * 32 + ((lane >= 16) ? 8 : 0);
#pragma unroll
    for (int i = 0; i < 16; ++i)
        dst[(size_t)idx * 16 + i] = (__bf16)W[(kofs + (i < 8 ? i : i + 8)) * Nout + n];
}

// ---------------------------------------------------------------------------
// K1: z = h @ W_node   [N x 128] x [128 x 64] -> z (f32) + z (bf16)
// one wave = 16 rows x 64 cols; 4 K-steps x 4 N-tiles
// ---------------------------------------------------------------------------
__global__ void node_embed_wmma(const float* __restrict__ h,
                                const __bf16* __restrict__ Wn_pk,
                                float* __restrict__ z,
                                __bf16* __restrict__ z_bf,
                                int nrows) {
    int wave = (int)((blockIdx.x * blockDim.x + threadIdx.x) >> 5);
    int lane = threadIdx.x & 31;
    int m0 = wave * 16;
    if (m0 >= nrows) return;                       // wave-uniform

    int khalf = (lane >= 16) ? 8 : 0;
    int arow  = m0 + (lane & 15);
    if (arow >= nrows) arow = nrows - 1;           // tail clamp (harmless dup)

    f32x8 zero = {};
    f32x8 acc[4] = {zero, zero, zero, zero};

#pragma unroll
    for (int ks = 0; ks < 4; ++ks) {
        int k0 = ks * 32;
        BF16Vec a;
        const float* ap = h + (size_t)arow * 128 + k0 + khalf;
        a.h[0] = cvt_bf8(ap);
        a.h[1] = cvt_bf8(ap + 16);
#pragma unroll
        for (int t = 0; t < 4; ++t) {
            BF16Vec b;
            const __bf16* bp = Wn_pk + (((size_t)ks * 4 + t) * 32 + lane) * 16;
            b.h[0] = load_bf8(bp);
            b.h[1] = load_bf8(bp + 8);
            acc[t] = __builtin_amdgcn_wmma_f32_16x16x32_bf16(
                false, a.v, false, b.v, (short)0, acc[t], false, false);
        }
    }

    int hi = (lane >= 16) ? 8 : 0;
    int ncol = lane & 15;
#pragma unroll
    for (int t = 0; t < 4; ++t) {
#pragma unroll
        for (int j = 0; j < 8; ++j) {
            int m = m0 + hi + j;
            if (m < nrows) {
                float val = acc[t][j];
                z[(size_t)m * 64 + t * 16 + ncol]    = val;
                z_bf[(size_t)m * 64 + t * 16 + ncol] = (__bf16)val;
            }
        }
    }
}

// ---------------------------------------------------------------------------
// K2: e = leakyrelu(zs.Wa0 + zd.Wa1 + ew.Wa2); atomic ordered-max per dst
// ---------------------------------------------------------------------------
__global__ void edge_attn_kernel(const float* __restrict__ z,
                                 const float* __restrict__ edge_w,
                                 const float* __restrict__ W_attn,
                                 const int* __restrict__ src,
                                 const int* __restrict__ dst,
                                 float* __restrict__ evals,
                                 unsigned* __restrict__ node_max,
                                 int nedges) {
    __shared__ float wa[160];
    for (int i = threadIdx.x; i < 160; i += blockDim.x) wa[i] = W_attn[i];
    __syncthreads();

    int e = blockIdx.x * blockDim.x + threadIdx.x;
    if (e >= nedges) return;
    int s = src[e], d = dst[e];
    float a = 0.0f;
#pragma unroll
    for (int k = 0; k < 64; k += 4) {
        float4 v = *(const float4*)(z + (size_t)s * 64 + k);
        a += v.x * wa[k] + v.y * wa[k + 1] + v.z * wa[k + 2] + v.w * wa[k + 3];
    }
#pragma unroll
    for (int k = 0; k < 64; k += 4) {
        float4 v = *(const float4*)(z + (size_t)d * 64 + k);
        a += v.x * wa[64 + k] + v.y * wa[65 + k] + v.z * wa[66 + k] + v.w * wa[67 + k];
    }
#pragma unroll
    for (int k = 0; k < 32; k += 4) {
        float4 v = *(const float4*)(edge_w + (size_t)e * 32 + k);
        a += v.x * wa[128 + k] + v.y * wa[129 + k] + v.z * wa[130 + k] + v.w * wa[131 + k];
    }
    float lr = (a > 0.0f) ? a : 0.1f * a;
    evals[e] = lr;
    atomicMax(&node_max[d], f32ord(lr));
}

// ---------------------------------------------------------------------------
// K3: ex = exp(e - max[dst]); atomic segment-sum + in-degree
// ---------------------------------------------------------------------------
__global__ void edge_exp_kernel(const unsigned* __restrict__ node_max,
                                const int* __restrict__ dst,
                                float* __restrict__ evals,
                                float* __restrict__ node_sum,
                                float* __restrict__ ndeg,
                                int nedges) {
    int e = blockIdx.x * blockDim.x + threadIdx.x;
    if (e >= nedges) return;
    int d = dst[e];
    float v = __expf(evals[e] - ordf32(node_max[d]));
    evals[e] = v;
    atomicAdd(&node_sum[d], v);
    atomicAdd(&ndeg[d], 1.0f);
}

// ---------------------------------------------------------------------------
// K4: t = [zs | ew] @ W_tonode ([E x 96] x [96 x 64]);
//     h_agg[dst] += alpha * t  (float atomics, L2-resident target)
// one wave = 32 edges (2 M-tiles share every B fetch); 3 Ksteps x 4 Ntiles x 2
// ---------------------------------------------------------------------------
__global__ void edge_tonode_wmma(const __bf16* __restrict__ z_bf,
                                 const float* __restrict__ edge_w,
                                 const __bf16* __restrict__ Wt_pk,
                                 const int* __restrict__ src,
                                 const int* __restrict__ dst,
                                 const float* __restrict__ ex,
                                 const float* __restrict__ node_sum,
                                 float* __restrict__ hagg,
                                 int nedges) {
    int wave = (int)((blockIdx.x * blockDim.x + threadIdx.x) >> 5);
    int lane = threadIdx.x & 31;
    int e0 = wave * 32;
    if (e0 >= nedges) return;

    int khalf = (lane >= 16) ? 8 : 0;
    int arow0 = e0 + (lane & 15);
    int arow1 = arow0 + 16;
    if (arow0 >= nedges) arow0 = nedges - 1;
    if (arow1 >= nedges) arow1 = nedges - 1;
    int s0 = src[arow0];
    int s1 = src[arow1];
    if (e0 + 32 < nedges) __builtin_prefetch(&edge_w[(size_t)(e0 + 32) * 32], 0, 1);

    f32x8 zero = {};
    f32x8 acc[2][4] = {{zero, zero, zero, zero}, {zero, zero, zero, zero}};

#pragma unroll
    for (int ks = 0; ks < 3; ++ks) {
        int k0 = ks * 32;
        BF16Vec a0, a1;
        if (k0 < 64) {
            const __bf16* p0 = z_bf + (size_t)s0 * 64 + k0 + khalf;
            const __bf16* p1 = z_bf + (size_t)s1 * 64 + k0 + khalf;
            a0.h[0] = load_bf8(p0);  a0.h[1] = load_bf8(p0 + 16);
            a1.h[0] = load_bf8(p1);  a1.h[1] = load_bf8(p1 + 16);
        } else {
            const float* p0 = edge_w + (size_t)arow0 * 32 + khalf;
            const float* p1 = edge_w + (size_t)arow1 * 32 + khalf;
            a0.h[0] = cvt_bf8(p0);   a0.h[1] = cvt_bf8(p0 + 16);
            a1.h[0] = cvt_bf8(p1);   a1.h[1] = cvt_bf8(p1 + 16);
        }
#pragma unroll
        for (int t = 0; t < 4; ++t) {
            BF16Vec b;
            const __bf16* bp = Wt_pk + (((size_t)ks * 4 + t) * 32 + lane) * 16;
            b.h[0] = load_bf8(bp);
            b.h[1] = load_bf8(bp + 8);
            acc[0][t] = __builtin_amdgcn_wmma_f32_16x16x32_bf16(
                false, a0.v, false, b.v, (short)0, acc[0][t], false, false);
            acc[1][t] = __builtin_amdgcn_wmma_f32_16x16x32_bf16(
                false, a1.v, false, b.v, (short)0, acc[1][t], false, false);
        }
    }

    int hi = (lane >= 16) ? 8 : 0;
    int ncol = lane & 15;
#pragma unroll
    for (int mt = 0; mt < 2; ++mt) {
#pragma unroll
        for (int j = 0; j < 8; ++j) {
            int eidx = e0 + mt * 16 + hi + j;
            if (eidx < nedges) {
                int d = dst[eidx];
                float alpha = ex[eidx] / node_sum[d];
#pragma unroll
                for (int t = 0; t < 4; ++t)
                    atomicAdd(&hagg[(size_t)d * 64 + t * 16 + ncol],
                              alpha * acc[mt][t][j]);
            }
        }
    }
}

// ---------------------------------------------------------------------------
// K5: h_new = indeg>0 ? h_agg : z   (in place in d_out) + bf16 copy
// ---------------------------------------------------------------------------
__global__ void finalize_h(const float* __restrict__ z,
                           const float* __restrict__ ndeg,
                           float* __restrict__ hnew,
                           __bf16* __restrict__ hnew_bf,
                           int nNodes) {
    int idx = blockIdx.x * blockDim.x + threadIdx.x;
    if (idx >= nNodes * 64) return;
    int node = idx >> 6;
    float v = (ndeg[node] > 0.0f) ? hnew[idx] : z[idx];
    hnew[idx]    = v;
    hnew_bf[idx] = (__bf16)v;
}

// ---------------------------------------------------------------------------
// K6: w_new = [h_new[src] | h_new[dst] | ew] @ W_edge ([E x 160] x [160 x 32])
// one wave = 32 edges; 5 Ksteps x 2 Ntiles x 2 M-tiles = 20 WMMAs
// ---------------------------------------------------------------------------
__global__ void edge_out_wmma(const __bf16* __restrict__ hnew_bf,
                              const float* __restrict__ edge_w,
                              const __bf16* __restrict__ We_pk,
                              const int* __restrict__ src,
                              const int* __restrict__ dst,
                              float* __restrict__ wout,
                              int nedges) {
    int wave = (int)((blockIdx.x * blockDim.x + threadIdx.x) >> 5);
    int lane = threadIdx.x & 31;
    int e0 = wave * 32;
    if (e0 >= nedges) return;

    int khalf = (lane >= 16) ? 8 : 0;
    int arow0 = e0 + (lane & 15);
    int arow1 = arow0 + 16;
    if (arow0 >= nedges) arow0 = nedges - 1;
    if (arow1 >= nedges) arow1 = nedges - 1;
    int s0 = src[arow0], d0 = dst[arow0];
    int s1 = src[arow1], d1 = dst[arow1];
    if (e0 + 32 < nedges) __builtin_prefetch(&edge_w[(size_t)(e0 + 32) * 32], 0, 1);

    f32x8 zero = {};
    f32x8 acc[2][2] = {{zero, zero}, {zero, zero}};

#pragma unroll
    for (int ks = 0; ks < 5; ++ks) {
        int k0 = ks * 32;
        BF16Vec a0, a1;
        if (k0 < 64) {
            const __bf16* p0 = hnew_bf + (size_t)s0 * 64 + k0 + khalf;
            const __bf16* p1 = hnew_bf + (size_t)s1 * 64 + k0 + khalf;
            a0.h[0] = load_bf8(p0);  a0.h[1] = load_bf8(p0 + 16);
            a1.h[0] = load_bf8(p1);  a1.h[1] = load_bf8(p1 + 16);
        } else if (k0 < 128) {
            const __bf16* p0 = hnew_bf + (size_t)d0 * 64 + (k0 - 64) + khalf;
            const __bf16* p1 = hnew_bf + (size_t)d1 * 64 + (k0 - 64) + khalf;
            a0.h[0] = load_bf8(p0);  a0.h[1] = load_bf8(p0 + 16);
            a1.h[0] = load_bf8(p1);  a1.h[1] = load_bf8(p1 + 16);
        } else {
            const float* p0 = edge_w + (size_t)arow0 * 32 + khalf;
            const float* p1 = edge_w + (size_t)arow1 * 32 + khalf;
            a0.h[0] = cvt_bf8(p0);   a0.h[1] = cvt_bf8(p0 + 16);
            a1.h[0] = cvt_bf8(p1);   a1.h[1] = cvt_bf8(p1 + 16);
        }
#pragma unroll
        for (int t = 0; t < 2; ++t) {
            BF16Vec b;
            const __bf16* bp = We_pk + (((size_t)ks * 2 + t) * 32 + lane) * 16;
            b.h[0] = load_bf8(bp);
            b.h[1] = load_bf8(bp + 8);
            acc[0][t] = __builtin_amdgcn_wmma_f32_16x16x32_bf16(
                false, a0.v, false, b.v, (short)0, acc[0][t], false, false);
            acc[1][t] = __builtin_amdgcn_wmma_f32_16x16x32_bf16(
                false, a1.v, false, b.v, (short)0, acc[1][t], false, false);
        }
    }

    int hi = (lane >= 16) ? 8 : 0;
    int ncol = lane & 15;
#pragma unroll
    for (int mt = 0; mt < 2; ++mt) {
#pragma unroll
        for (int j = 0; j < 8; ++j) {
            int eidx = e0 + mt * 16 + hi + j;
            if (eidx < nedges) {
#pragma unroll
                for (int t = 0; t < 2; ++t)
                    wout[(size_t)eidx * 32 + t * 16 + ncol] = acc[mt][t][j];
            }
        }
    }
}

// ---------------------------------------------------------------------------
// host driver
// ---------------------------------------------------------------------------
extern "C" void kernel_launch(void* const* d_in, const int* in_sizes, int n_in,
                              void* d_out, int out_size, void* d_ws, size_t ws_size,
                              hipStream_t stream) {
    const float* h       = (const float*)d_in[0];   // [N,128]
    const float* edge_w  = (const float*)d_in[1];   // [E,32]
    const float* W_node  = (const float*)d_in[2];   // [128,64]
    const float* W_attn  = (const float*)d_in[3];   // [160,1]
    const float* W_ton   = (const float*)d_in[4];   // [96,64]
    const float* W_edge  = (const float*)d_in[5];   // [160,32]
    const int*   src     = (const int*)d_in[6];     // [E]
    const int*   dst     = (const int*)d_in[7];     // [E]

    const int N = in_sizes[0] / 128;
    const int E = in_sizes[6];

    float* h_new = (float*)d_out;                   // [N,64]
    float* w_new = (float*)d_out + (size_t)N * 64;  // [E,32]

    // workspace carve-out (256B aligned)
    char* ws = (char*)d_ws;
    size_t off = 0;
    auto take = [&](size_t bytes) -> char* {
        char* p = ws + off;
        off = (off + bytes + 255) & ~(size_t)255;
        return p;
    };
    float*    z       = (float*)take((size_t)N * 64 * 4);
    __bf16*   z_bf    = (__bf16*)take((size_t)N * 64 * 2);
    __bf16*   hnew_bf = (__bf16*)take((size_t)N * 64 * 2);
    float*    evals   = (float*)take((size_t)E * 4);
    unsigned* nmax    = (unsigned*)take((size_t)N * 4);
    float*    nsum    = (float*)take((size_t)N * 4);
    float*    ndeg    = (float*)take((size_t)N * 4);
    __bf16*   Wn_pk   = (__bf16*)take(128 * 64 * 2);
    __bf16*   Wt_pk   = (__bf16*)take(96 * 64 * 2);
    __bf16*   We_pk   = (__bf16*)take(160 * 32 * 2);
    (void)ws_size; (void)n_in; (void)out_size;

    const int T = 256;

    // K0: init accumulators
    init_kernel<<<(N * 64 + T - 1) / T, T, 0, stream>>>(h_new, nmax, nsum, ndeg, N);

    // K0b: pack weights to WMMA-B bf16 layout
    pack_weights<<<2, T, 0, stream>>>(W_node, Wn_pk, 128, 64);  // 512 groups
    pack_weights<<<2, T, 0, stream>>>(W_ton,  Wt_pk,  96, 64);  // 384 groups
    pack_weights<<<2, T, 0, stream>>>(W_edge, We_pk, 160, 32);  // 320 groups

    // K1: node embed (WMMA)
    {
        int tiles  = (N + 15) / 16;
        int blocks = (tiles + 7) / 8;               // 8 waves / block
        node_embed_wmma<<<blocks, T, 0, stream>>>(h, Wn_pk, z, z_bf, N);
    }

    // K2: attention logits + segment max
    edge_attn_kernel<<<(E + T - 1) / T, T, 0, stream>>>(z, edge_w, W_attn, src, dst,
                                                        evals, nmax, E);

    // K3: exp + segment sum + indeg
    edge_exp_kernel<<<(E + T - 1) / T, T, 0, stream>>>(nmax, dst, evals, nsum, ndeg, E);

    // K4: to-node GEMM (WMMA) + alpha-weighted aggregation
    {
        int tiles  = (E + 31) / 32;
        int blocks = (tiles + 7) / 8;
        edge_tonode_wmma<<<blocks, T, 0, stream>>>(z_bf, edge_w, Wt_pk, src, dst,
                                                   evals, nsum, h_new, E);
    }

    // K5: zero-indegree fallback + bf16 copy of h_new
    finalize_h<<<(N * 64 + T - 1) / T, T, 0, stream>>>(z, ndeg, h_new, hnew_bf, N);

    // K6: edge output GEMM (WMMA)
    {
        int tiles  = (E + 31) / 32;
        int blocks = (tiles + 7) / 8;
        edge_out_wmma<<<blocks, T, 0, stream>>>(hnew_bf, edge_w, We_pk, src, dst,
                                                w_new, E);
    }
}
